// SerialBlock_full_85100482003057
// MI455X (gfx1250) — compile-verified, matrix operand order checked by reading, not run on hardware
//
#include <hip/hip_runtime.h>
#include <math.h>

// ---------------------------------------------------------------------------
// CoaT-style serial-block model forward for MI455X (gfx1250, wave32, WMMA).
// All GEMMs (qkv/proj/fc1/fc2/patch-embed-im2col/attention einsums) run on
// v_wmma_f32_16x16x32_f16, 32x32 output tile per wave (2x2 WMMA fragments).
// A operand is always K-contiguous (stride 1) => immediate-offset loads.
// ---------------------------------------------------------------------------

typedef _Float16 f16;
typedef __attribute__((ext_vector_type(16))) _Float16 v16h;
typedef __attribute__((ext_vector_type(8)))  float    v8f;

#define NUMH 8

struct GemmP {
  const float* A;    // A operand (f32 in memory), A[m][k] = A[m*lda + k]
  const float* Bm;   // B operand: B[n][k] = Bm[n*ldb + k*sb] (C = A * B^T)
  const float* bias; // per-n bias or nullptr
  const float* Res;  // residual added after activation, or nullptr
  float*       C;    // output
  long lda;
  long ldb, sb;
  long ldc;          // C[m][n] = C[m*ldc + n]
  long ldres;        // Res[m][n] = Res[m*ldres + n]
  long offAo, offAi; // batch offsets: z = bo*innerB + bi
  long offBo, offBi;
  long offCo, offCi;
  long offRo, offRi;
  int  innerB;
  int  M, N, K;
  float alpha;       // C = act(alpha*acc + bias) + Res
  int  act;          // 0 = none, 1 = exact GELU
  int  amode;        // 0 strided, 1 im2col NCHW (ks=16), 2 im2col tokens (ks=2)
  int  Cin, inH, inW, outW; // im2col params (stride == kernel)
};

__device__ __forceinline__ int imin(int a, int b) { return a < b ? a : b; }

// im2col element address for A (AMODE 1: raw NCHW, ks=16; AMODE 2: token layout, ks=2)
template <int AMODE>
__device__ __forceinline__ long a_im2col(const GemmP& p, int oyk, int oxk, int kk) {
  if (AMODE == 1) {
    int ci = kk >> 8, rem = kk & 255, ky = rem >> 4, kx = rem & 15;
    return ((long)ci * p.inH + (oyk + ky)) * p.inW + (oxk + kx);
  } else {
    int ci = kk >> 2, rem = kk & 3, ky = rem >> 1, kx = rem & 1;
    return (long)(1 + (oyk + ky) * p.inW + (oxk + kx)) * p.Cin + ci;
  }
}

template <int AMODE, bool SB1>
__global__ __launch_bounds__(128) void gemm_wmma(GemmP p) {
  const int wave = threadIdx.x >> 5;
  const int lane = threadIdx.x & 31;
  const int r    = lane & 15;
  const int hi   = lane >> 4;

  const int mtiles = (p.M + 31) >> 5;
  const int ntiles = (p.N + 31) >> 5;
  const long tile  = (long)blockIdx.x * 4 + wave;
  if (tile >= (long)mtiles * ntiles) return;     // wave-uniform
  const int mt = (int)(tile / ntiles);
  const int nt = (int)(tile % ntiles);

  const int z  = blockIdx.z;
  const int bo = z / p.innerB;
  const int bi = z - bo * p.innerB;

  const float* Ab = p.A  + (long)bo * p.offAo + (long)bi * p.offAi;
  const float* Bb = p.Bm + (long)bo * p.offBo + (long)bi * p.offBi;

  // Clamp rows/cols into range: out-of-range lanes fetch valid (finite) data;
  // their outputs are simply not stored. No predication in the hot loop.
  const int row0 = imin(mt * 32 + r,      p.M - 1);
  const int row1 = imin(mt * 32 + 16 + r, p.M - 1);
  const int col0 = imin(nt * 32 + r,      p.N - 1);
  const int col1 = imin(nt * 32 + 16 + r, p.N - 1);

  // Base pointers (amode 0) / row decompositions (im2col modes).
  const float* pA0 = Ab;
  const float* pA1 = Ab;
  int oy0 = 0, ox0 = 0, oy1 = 0, ox1 = 0;
  if (AMODE == 0) {
    pA0 = Ab + (long)row0 * p.lda;
    pA1 = Ab + (long)row1 * p.lda;
  } else {
    const int ks = (AMODE == 1) ? 16 : 2;
    oy0 = (row0 / p.outW) * ks; ox0 = (row0 % p.outW) * ks;
    oy1 = (row1 / p.outW) * ks; ox1 = (row1 % p.outW) * ks;
  }
  const float* pB0 = Bb + (long)col0 * p.ldb;
  const float* pB1 = Bb + (long)col1 * p.ldb;
  const long sb = p.sb;

  v8f acc00 = {0.f,0.f,0.f,0.f,0.f,0.f,0.f,0.f};
  v8f acc01 = acc00, acc10 = acc00, acc11 = acc00;

  const int Kmain = p.K & ~31;
  int k0 = 0;
  for (; k0 < Kmain; k0 += 32) {
    v16h a0f, a1f, b0f, b1f;
#pragma unroll
    for (int j = 0; j < 8; ++j) {
      // f16 A 16x32 fragment K-mapping (ISA 7.12.2); B symmetric (M->N).
      const int kp = ((j >> 2) << 4) + (hi << 3) + ((j & 3) << 1);
      const int kk = k0 + kp;
      float a00, a01, a10, a11;
      if (AMODE == 0) {
        a00 = pA0[kk]; a01 = pA0[kk + 1];   // stride-1 => immediate offsets
        a10 = pA1[kk]; a11 = pA1[kk + 1];
      } else {
        a00 = Ab[a_im2col<AMODE>(p, oy0, ox0, kk)];
        a01 = Ab[a_im2col<AMODE>(p, oy0, ox0, kk + 1)];
        a10 = Ab[a_im2col<AMODE>(p, oy1, ox1, kk)];
        a11 = Ab[a_im2col<AMODE>(p, oy1, ox1, kk + 1)];
      }
      float b00, b01, b10, b11;
      if (SB1) {
        b00 = pB0[kk]; b01 = pB0[kk + 1];
        b10 = pB1[kk]; b11 = pB1[kk + 1];
      } else {
        const long f0 = (long)kk * sb, f1 = f0 + sb;
        b00 = pB0[f0]; b01 = pB0[f1];
        b10 = pB1[f0]; b11 = pB1[f1];
      }
      a0f[2*j] = (f16)a00; a0f[2*j+1] = (f16)a01;
      a1f[2*j] = (f16)a10; a1f[2*j+1] = (f16)a11;
      b0f[2*j] = (f16)b00; b0f[2*j+1] = (f16)b01;
      b1f[2*j] = (f16)b10; b1f[2*j+1] = (f16)b11;
    }
    acc00 = __builtin_amdgcn_wmma_f32_16x16x32_f16(false, a0f, false, b0f, (short)0, acc00, false, false);
    acc01 = __builtin_amdgcn_wmma_f32_16x16x32_f16(false, a0f, false, b1f, (short)0, acc01, false, false);
    acc10 = __builtin_amdgcn_wmma_f32_16x16x32_f16(false, a1f, false, b0f, (short)0, acc10, false, false);
    acc11 = __builtin_amdgcn_wmma_f32_16x16x32_f16(false, a1f, false, b1f, (short)0, acc11, false, false);
  }

  if (k0 < p.K) {
    // Tail: clamp k for addressing, zero out-of-range values via cndmask
    // (both A and B: WMMA defines 0 x Inf -> NaN, so B must be zeroed too).
    v16h a0f, a1f, b0f, b1f;
#pragma unroll
    for (int j = 0; j < 8; ++j) {
      const int kp  = ((j >> 2) << 4) + (hi << 3) + ((j & 3) << 1);
      const int kk0 = k0 + kp, kk1 = kk0 + 1;
      const int kc0 = imin(kk0, p.K - 1), kc1 = imin(kk1, p.K - 1);
      const bool v0 = kk0 < p.K, v1 = kk1 < p.K;
      float a00, a01, a10, a11;
      if (AMODE == 0) {
        a00 = pA0[kc0]; a01 = pA0[kc1];
        a10 = pA1[kc0]; a11 = pA1[kc1];
      } else {
        a00 = Ab[a_im2col<AMODE>(p, oy0, ox0, kc0)];
        a01 = Ab[a_im2col<AMODE>(p, oy0, ox0, kc1)];
        a10 = Ab[a_im2col<AMODE>(p, oy1, ox1, kc0)];
        a11 = Ab[a_im2col<AMODE>(p, oy1, ox1, kc1)];
      }
      float b00, b01, b10, b11;
      if (SB1) {
        b00 = pB0[kc0]; b01 = pB0[kc1];
        b10 = pB1[kc0]; b11 = pB1[kc1];
      } else {
        b00 = pB0[(long)kc0 * sb]; b01 = pB0[(long)kc1 * sb];
        b10 = pB1[(long)kc0 * sb]; b11 = pB1[(long)kc1 * sb];
      }
      a0f[2*j] = (f16)(v0 ? a00 : 0.f); a0f[2*j+1] = (f16)(v1 ? a01 : 0.f);
      a1f[2*j] = (f16)(v0 ? a10 : 0.f); a1f[2*j+1] = (f16)(v1 ? a11 : 0.f);
      b0f[2*j] = (f16)(v0 ? b00 : 0.f); b0f[2*j+1] = (f16)(v1 ? b01 : 0.f);
      b1f[2*j] = (f16)(v0 ? b10 : 0.f); b1f[2*j+1] = (f16)(v1 ? b11 : 0.f);
    }
    acc00 = __builtin_amdgcn_wmma_f32_16x16x32_f16(false, a0f, false, b0f, (short)0, acc00, false, false);
    acc01 = __builtin_amdgcn_wmma_f32_16x16x32_f16(false, a0f, false, b1f, (short)0, acc01, false, false);
    acc10 = __builtin_amdgcn_wmma_f32_16x16x32_f16(false, a1f, false, b0f, (short)0, acc10, false, false);
    acc11 = __builtin_amdgcn_wmma_f32_16x16x32_f16(false, a1f, false, b1f, (short)0, acc11, false, false);
  }

  // ---- epilogue ----
  float*       Cb = p.C + (long)bo * p.offCo + (long)bi * p.offCi;
  const float* Rb = p.Res ? (p.Res + (long)bo * p.offRo + (long)bi * p.offRi) : nullptr;

#pragma unroll
  for (int half = 0; half < 2; ++half) {
    const int n = nt * 32 + half * 16 + r;
    const v8f* accs0 = half ? &acc01 : &acc00;
    const v8f* accs1 = half ? &acc11 : &acc10;
    if (n < p.N) {
      const float bv = p.bias ? p.bias[n] : 0.f;
#pragma unroll
      for (int v = 0; v < 8; ++v) {
        // C/D layout: VGPR v -> M = v (lanes 0-15), v+8 (lanes 16-31)
        int m0 = mt * 32 + v + hi * 8;
        int m1 = m0 + 16;
        float val0 = (*accs0)[v] * p.alpha + bv;
        float val1 = (*accs1)[v] * p.alpha + bv;
        if (p.act == 1) {
          val0 = 0.5f * val0 * (1.f + erff(val0 * 0.70710678118f));
          val1 = 0.5f * val1 * (1.f + erff(val1 * 0.70710678118f));
        }
        if (m0 < p.M) {
          if (Rb) val0 += Rb[(long)m0 * p.ldres + n];
          Cb[(long)m0 * p.ldc + n] = val0;
        }
        if (m1 < p.M) {
          if (Rb) val1 += Rb[(long)m1 * p.ldres + n];
          Cb[(long)m1 * p.ldc + n] = val1;
        }
      }
    }
  }
}

// ------------------------- LayerNorm (one wave / row) -----------------------
__global__ __launch_bounds__(128) void ln_kernel(const float* in, float* out,
                                                 const float* g, const float* b,
                                                 int Bb, int NP, int Nfull, int start,
                                                 int C, float eps) {
  int wave = threadIdx.x >> 5, lane = threadIdx.x & 31;
  long row = (long)blockIdx.x * 4 + wave;
  if (row >= (long)Bb * NP) return;
  int bb = (int)(row / NP), rr = (int)(row % NP);
  const float* x = in  + ((long)bb * Nfull + start + rr) * C;
  float*       y = out + ((long)bb * Nfull + start + rr) * C;
  float v[12];
  int cnt = 0; float sum = 0.f;
  for (int c = lane; c < C; c += 32) { v[cnt] = x[c]; sum += v[cnt]; ++cnt; }
  for (int o = 16; o; o >>= 1) sum += __shfl_xor(sum, o, 32);
  float mean = sum / (float)C;
  float var = 0.f;
  for (int i = 0; i < cnt; ++i) { float d = v[i] - mean; var += d * d; }
  for (int o = 16; o; o >>= 1) var += __shfl_xor(var, o, 32);
  var /= (float)C;
  float inv = rsqrtf(var + eps);
  cnt = 0;
  for (int c = lane; c < C; c += 32) { y[c] = (v[cnt] - mean) * inv * g[c] + b[c]; ++cnt; }
}

// -------- softmax over token axis of K, writing K^T (B, h, Ch, N) -----------
__global__ void softmax_k(const float* qkv, float* ksmT,
                          int Bb, int Ch, int N, int threeC) {
  long idx = (long)blockIdx.x * blockDim.x + threadIdx.x;
  long total = (long)Bb * NUMH * Ch;
  if (idx >= total) return;
  int ch   = (int)(idx % Ch);
  int head = (int)((idx / Ch) % NUMH);
  int bb   = (int)(idx / ((long)Ch * NUMH));
  const float* base = qkv + (long)bb * N * threeC + (long)(NUMH + head) * Ch + ch;
  float mx = -3.0e38f;
  for (int n = 0; n < N; ++n) mx = fmaxf(mx, base[(long)n * threeC]);
  float s = 0.f;
  for (int n = 0; n < N; ++n) s += expf(base[(long)n * threeC] - mx);
  float inv = 1.f / s;
  float* o = ksmT + idx * (long)N;
  for (int n = 0; n < N; ++n) o[n] = expf(base[(long)n * threeC] - mx) * inv;
}

// -------- CPE: depthwise 3x3 conv on image tokens + identity; cls copied ----
__global__ void cpe_kernel(const float* xin, float* xout, const float* w,
                           const float* bias, int Bb, int N, int C, int H, int W) {
  long idx = (long)blockIdx.x * blockDim.x + threadIdx.x;
  long total = (long)Bb * N * C;
  if (idx >= total) return;
  int c  = (int)(idx % C);
  long t = idx / C;
  int n  = (int)(t % N);
  int bb = (int)(t / N);
  const float* xb = xin + (long)bb * N * C;
  float xv = xb[(long)n * C + c];
  if (n == 0) { xout[idx] = xv; return; }
  int p = n - 1, y = p / W, x = p - y * W;
  float s = bias[c] + xv;
  const float* wc = w + (long)c * 9;
  for (int dy = 0; dy < 3; ++dy) {
    int yy = y + dy - 1; if (yy < 0 || yy >= H) continue;
    for (int dx = 0; dx < 3; ++dx) {
      int xx = x + dx - 1; if (xx < 0 || xx >= W) continue;
      s += xb[(long)(1 + yy * W + xx) * C + c] * wc[dy * 3 + dx];
    }
  }
  xout[idx] = s;
}

// -------- CRPE: grouped depthwise conv on V image, -> cv (B, h, N, Ch) ------
__global__ void crpe_conv(const float* qkv, float* cv,
                          const float* w0, const float* b0,
                          const float* w1, const float* b1,
                          const float* w2, const float* b2,
                          int Bb, int Ch, int N, int H, int W, int threeC) {
  long idx = (long)blockIdx.x * blockDim.x + threadIdx.x;
  long total = (long)Bb * NUMH * H * W * Ch;
  if (idx >= total) return;
  int ch = (int)(idx % Ch);
  long t = idx / Ch;
  int hw = (int)(t % (H * W)); t /= (H * W);
  int head = (int)(t % NUMH);
  int bb   = (int)(t / NUMH);
  int y = hw / W, x = hw - y * W;
  const float* w; const float* bs; int win, hb;
  if (head < 2)      { w = w0; bs = b0; win = 3; hb = 0; }
  else if (head < 5) { w = w1; bs = b1; win = 5; hb = 2; }
  else               { w = w2; bs = b2; win = 7; hb = 5; }
  int cg = (head - hb) * Ch + ch;
  const float* wc = w + (long)cg * win * win;
  int pad = win / 2;
  const float* vb = qkv + (long)bb * N * threeC + (long)(2 * NUMH + head) * Ch + ch;
  float s = bs[cg];
  for (int dy = 0; dy < win; ++dy) {
    int yy = y + dy - pad; if (yy < 0 || yy >= H) continue;
    for (int dx = 0; dx < win; ++dx) {
      int xx = x + dx - pad; if (xx < 0 || xx >= W) continue;
      s += vb[(long)(1 + yy * W + xx) * threeC] * wc[dy * win + dx];
    }
  }
  cv[(((long)(bb * NUMH + head) * N) + 1 + hw) * Ch + ch] = s;
}

// -------- combine: attn[b,n,h*Ch+ch] = fa + (n>0 ? q*cv : 0) ----------
__global__ void combine_attn(const float* fa, const float* qkv, const float* cv,
                             float* attn, int Bb, int Ch, int N, int C, int threeC) {
  long idx = (long)blockIdx.x * blockDim.x + threadIdx.x;
  long total = (long)Bb * N * C;
  if (idx >= total) return;
  int c  = (int)(idx % C);
  long t = idx / C;
  int n  = (int)(t % N);
  int bb = (int)(t / N);
  int head = c / Ch, ch = c - head * Ch;
  long bh = (long)(bb * NUMH + head);
  float val = fa[(bh * N + n) * Ch + ch];
  if (n > 0) {
    float q = qkv[((long)bb * N + n) * threeC + head * Ch + ch];
    val += q * cv[(bh * N + n) * Ch + ch];
  }
  attn[idx] = val;
}

__global__ void cls_fill(float* tok, const float* cls, int Bb, int N, int C) {
  int idx = blockIdx.x * blockDim.x + threadIdx.x;
  if (idx >= Bb * C) return;
  int c = idx % C, bb = idx / C;
  tok[(long)bb * N * C + c] = cls[c];
}

// ---------------------------------------------------------------------------
static inline void launch_gemm(hipStream_t st, GemmP& p, int batches) {
  int mtiles = (p.M + 31) / 32, ntiles = (p.N + 31) / 32;
  long tiles = (long)mtiles * ntiles;
  int bx = (int)((tiles + 3) / 4);
  dim3 g(bx, 1, batches), b(128, 1, 1);
  if (p.amode == 0) {
    if (p.sb == 1) gemm_wmma<0, true ><<<g, b, 0, st>>>(p);
    else           gemm_wmma<0, false><<<g, b, 0, st>>>(p);
  } else if (p.amode == 1) {
    gemm_wmma<1, true><<<g, b, 0, st>>>(p);
  } else {
    gemm_wmma<2, true><<<g, b, 0, st>>>(p);
  }
}

static inline int ceildiv(long a, long b) { return (int)((a + b - 1) / b); }

extern "C" void kernel_launch(void* const* d_in, const int* in_sizes, int n_in,
                              void* d_out, int out_size, void* d_ws, size_t ws_size,
                              hipStream_t stream) {
  (void)ws_size; (void)out_size;
  static const int EMB[4]   = {64, 128, 256, 320};
  static const int RATIO[4] = {8, 8, 4, 4};

  // Locate x by its unique flat size; params (148 leaves) keep relative order.
  const int XSIZE = 32 * 3 * 384 * 384;
  int xi = 0;
  for (int i = 0; i < n_in; ++i) if (in_sizes[i] == XSIZE) { xi = i; break; }
  const float* x = (const float*)d_in[xi];
  const float* prm[148];
  {
    int j = 0;
    for (int i = 0; i < n_in && j < 148; ++i) { if (i == xi) continue; prm[j++] = (const float*)d_in[i]; }
  }
  // Per-stage leaf order (jax tree flatten, sorted keys), 37 leaves/stage:
  //  blk*12: [fc1_b, fc1_w, fc2_b, fc2_w, n1_b, n1_g, n2_b, n2_g, proj_b,
  //           proj_w, qkv_b, qkv_w]   (blocks 0 then 1)
  //  24:cls  25:cpe_b 26:cpe_w 27..32: crpe (w0,b0,w1,b1,w2,b2)
  //  33:patch_b 34:patch_ln_b 35:patch_ln_g 36:patch_w

  // Workspace carve (f32).
  float* wsp = (float*)d_ws;
  long off = 0;
  auto alloc = [&](long nf) { float* p = wsp + off; off += nf; return p; };
  const long MAXTOK = 32L * 577 * 64;
  float* bufX = alloc(MAXTOK);
  float* bufY = alloc(MAXTOK);
  float* bufL = alloc(MAXTOK);
  float* qkvb = alloc(3 * MAXTOK);
  float* ksmT = alloc(MAXTOK);
  float* kvb  = alloc(32L * NUMH * 40 * 40);
  float* fab  = alloc(MAXTOK);
  float* cvb  = alloc(MAXTOK);
  float* attb = alloc(MAXTOK);
  float* hidb = alloc(32L * 577 * 512);

  const int Bb = 32;
  int prevH = 0, prevW = 0, prevN = 0, prevC = 0;

  for (int s = 0; s < 4; ++s) {
    const int C = EMB[s], Ch = C / NUMH, hid = C * RATIO[s], threeC = 3 * C;
    const int H = (s == 0) ? 24 : prevH / 2;
    const int W = (s == 0) ? 24 : prevW / 2;
    const int HW = H * W, N = HW + 1;
    const float* const* P = prm + s * 37;

    // ---- patch embed: im2col GEMM -> bufY rows 1..HW, then LN, then cls ----
    {
      GemmP g = {};
      if (s == 0) {
        g.A = x; g.amode = 1; g.Cin = 3; g.inH = 384; g.inW = 384;
        g.outW = W; g.offAo = 3L * 384 * 384; g.K = 3 * 16 * 16;
      } else {
        g.A = bufX; g.amode = 2; g.Cin = prevC; g.inH = prevH; g.inW = prevW;
        g.outW = W; g.offAo = (long)prevN * prevC; g.K = 4 * prevC;
      }
      g.Bm = P[36]; g.ldb = g.K; g.sb = 1;
      g.bias = P[33];
      g.C = bufY + C;  g.ldc = C; g.offCo = (long)N * C;
      g.innerB = 1; g.M = HW; g.N = C; g.alpha = 1.f;
      launch_gemm(stream, g, Bb);
    }
    ln_kernel<<<ceildiv((long)Bb * HW, 4), 128, 0, stream>>>(
        bufY, bufY, P[35], P[34], Bb, HW, N, 1, C, 1e-5f);
    cls_fill<<<ceildiv(Bb * C, 256), 256, 0, stream>>>(bufY, P[24], Bb, N, C);
    { float* t = bufX; bufX = bufY; bufY = t; }  // x now in bufX

    const float scale = 1.f / sqrtf((float)Ch);

    for (int blk = 0; blk < 2; ++blk) {
      const float* const* bp = P + blk * 12;
      const float* fc1_b = bp[0];  const float* fc1_w = bp[1];
      const float* fc2_b = bp[2];  const float* fc2_w = bp[3];
      const float* n1_b  = bp[4];  const float* n1_g  = bp[5];
      const float* n2_b  = bp[6];  const float* n2_g  = bp[7];
      const float* prj_b = bp[8];  const float* prj_w = bp[9];
      const float* qkv_b = bp[10]; const float* qkv_w = bp[11];

      // 1) CPE: bufX -> bufY (x becomes bufY)
      cpe_kernel<<<ceildiv((long)Bb * N * C, 256), 256, 0, stream>>>(
          bufX, bufY, P[26], P[25], Bb, N, C, H, W);
      // 2) LN1: bufY -> bufL
      ln_kernel<<<ceildiv((long)Bb * N, 4), 128, 0, stream>>>(
          bufY, bufL, n1_g, n1_b, Bb, N, N, 0, C, 1e-6f);
      // 3) qkv GEMM
      {
        GemmP g = {};
        g.A = bufL; g.lda = C;
        g.Bm = qkv_w; g.ldb = C; g.sb = 1;
        g.bias = qkv_b;
        g.C = qkvb; g.ldc = threeC;
        g.innerB = 1; g.M = Bb * N; g.N = threeC; g.K = C; g.alpha = 1.f;
        launch_gemm(stream, g, 1);
      }
      // 4) softmax over tokens of K -> ksmT (B, h, Ch, N)
      softmax_k<<<ceildiv((long)Bb * NUMH * Ch, 256), 256, 0, stream>>>(
          qkvb, ksmT, Bb, Ch, N, threeC);
      // 5) kv = ksm^T @ v  (per b,h): (Ch x N)(N x Ch)
      {
        GemmP g = {};
        g.A = ksmT; g.lda = N;
        g.offAo = (long)NUMH * Ch * N; g.offAi = (long)Ch * N;
        g.Bm = qkvb + 2 * NUMH * Ch; g.ldb = 1; g.sb = threeC;
        g.offBo = (long)N * threeC;  g.offBi = Ch;
        g.C = kvb; g.ldc = Ch;
        g.offCo = (long)NUMH * Ch * Ch; g.offCi = (long)Ch * Ch;
        g.innerB = NUMH; g.M = Ch; g.N = Ch; g.K = N; g.alpha = 1.f;
        launch_gemm(stream, g, Bb * NUMH);
      }
      // 6) fa = scale * (q @ kv)  (per b,h): (N x Ch)(Ch x Ch)
      {
        GemmP g = {};
        g.A = qkvb; g.lda = threeC;
        g.offAo = (long)N * threeC; g.offAi = Ch;
        g.Bm = kvb; g.ldb = 1; g.sb = Ch;   // B[n][k] = kv[k][n]
        g.offBo = (long)NUMH * Ch * Ch; g.offBi = (long)Ch * Ch;
        g.C = fab; g.ldc = Ch;
        g.offCo = (long)NUMH * N * Ch; g.offCi = (long)N * Ch;
        g.innerB = NUMH; g.M = N; g.N = Ch; g.K = Ch; g.alpha = scale;
        launch_gemm(stream, g, Bb * NUMH);
      }
      // 7) CRPE depthwise convs on V -> cvb
      crpe_conv<<<ceildiv((long)Bb * NUMH * HW * Ch, 256), 256, 0, stream>>>(
          qkvb, cvb, P[27], P[28], P[29], P[30], P[31], P[32],
          Bb, Ch, N, H, W, threeC);
      // 8) attn = fa + q * cv (cls row gets fa only)
      combine_attn<<<ceildiv((long)Bb * N * C, 256), 256, 0, stream>>>(
          fab, qkvb, cvb, attb, Bb, Ch, N, C, threeC);
      // 9) proj GEMM + residual(bufY) -> bufX
      {
        GemmP g = {};
        g.A = attb; g.lda = C;
        g.Bm = prj_w; g.ldb = C; g.sb = 1;
        g.bias = prj_b;
        g.Res = bufY; g.ldres = C;
        g.C = bufX; g.ldc = C;
        g.innerB = 1; g.M = Bb * N; g.N = C; g.K = C; g.alpha = 1.f;
        launch_gemm(stream, g, 1);
      }
      // 10) LN2: bufX -> bufL
      ln_kernel<<<ceildiv((long)Bb * N, 4), 128, 0, stream>>>(
          bufX, bufL, n2_g, n2_b, Bb, N, N, 0, C, 1e-6f);
      // 11) fc1 + GELU -> hidb
      {
        GemmP g = {};
        g.A = bufL; g.lda = C;
        g.Bm = fc1_w; g.ldb = C; g.sb = 1;
        g.bias = fc1_b; g.act = 1;
        g.C = hidb; g.ldc = hid;
        g.innerB = 1; g.M = Bb * N; g.N = hid; g.K = C; g.alpha = 1.f;
        launch_gemm(stream, g, 1);
      }
      // 12) fc2 + residual(bufX) -> bufX
      {
        GemmP g = {};
        g.A = hidb; g.lda = hid;
        g.Bm = fc2_w; g.ldb = hid; g.sb = 1;
        g.bias = fc2_b;
        g.Res = bufX; g.ldres = C;
        g.C = bufX; g.ldc = C;
        g.innerB = 1; g.M = Bb * N; g.N = C; g.K = hid; g.alpha = 1.f;
        launch_gemm(stream, g, 1);
      }
    }

    prevH = H; prevW = W; prevN = N; prevC = C;
  }

  // Final x is (32, 10, 320) in bufX -> d_out
  hipMemcpyAsync(d_out, bufX, (size_t)(32L * 10 * 320) * sizeof(float),
                 hipMemcpyDeviceToDevice, stream);
}
